// SelfAttentionHead_35562329210956
// MI455X (gfx1250) — matrix operand order repaired and checked by reading, not compiled, not run on hardware
//
#include <hip/hip_runtime.h>
#include <hip/hip_bf16.h>

// ---------------------------------------------------------------------------
// Self-attention head, MI455X (gfx1250), wave32 + WMMA f16 (f32 accumulate).
//   x : [B=4, S=4096, E=768] fp32      Wk/Wq/Wv : [E, H=64] fp32
//   out = softmax((xWq)(xWk)^T / sqrt(E)) (xWv)   -> [B, S, H] fp32
// Pipeline: fp16 weight transpose (+Q scale fold) -> fp16 QKV projection via
// WMMA -> flash attention: async double-buffered K/V staging into LDS
// (GLOBAL_LOAD_ASYNC_TO_LDS_B128 when available), WMMA QK^T and PV, online
// softmax in fp32.
// ---------------------------------------------------------------------------

#define EMBED 768
#define HEADS 64
#define SEQ   4096
#define BATCH 4
#define NCHUNK (SEQ / 64)

#ifndef __has_builtin
#define __has_builtin(x) 0
#endif

#if defined(__AMDGCN__) && \
    __has_builtin(__builtin_amdgcn_global_load_async_to_lds_b128) && \
    __has_builtin(__builtin_amdgcn_s_wait_asynccnt)
#define USE_ASYNC_STAGE 1
#else
#define USE_ASYNC_STAGE 0
#endif

typedef __attribute__((ext_vector_type(16))) _Float16 v16h;
typedef __attribute__((ext_vector_type(8)))  _Float16 v8h;
typedef __attribute__((ext_vector_type(8)))  float    v8f;
typedef __attribute__((ext_vector_type(4)))  float    v4f;
typedef __attribute__((__vector_size__(4 * sizeof(int)))) int i128_t;

#define AS_GLOBAL __attribute__((address_space(1)))
#define AS_LDS    __attribute__((address_space(3)))

static __device__ __forceinline__ v16h pack16(v8h lo, v8h hi) {
    v16h r;
#pragma unroll
    for (int i = 0; i < 8; ++i) { r[i] = lo[i]; r[i + 8] = hi[i]; }
    return r;
}

// A-fragment (16x32, f16, MxK): lane holds row M=lane&15, K runs
// [g*8, g*8+7] and [g*8+16, g*8+23] with g = lane>>4 (ISA 7.12.2).
static __device__ __forceinline__ v16h load_a_h(const _Float16* base, int ld, int lane) {
    const int m = lane & 15, g = lane >> 4;
    const _Float16* p = base + m * ld + g * 8;
    return pack16(*(const v8h*)p, *(const v8h*)(p + 16));
}

// Same A-fragment sourced from fp32, converted inline.
static __device__ __forceinline__ v16h load_a_f32(const float* base, int ld, int lane) {
    const int m = lane & 15, g = lane >> 4;
    const float* p = base + m * ld + g * 8;
    v4f x0 = *(const v4f*)(p);
    v4f x1 = *(const v4f*)(p + 4);
    v4f y0 = *(const v4f*)(p + 16);
    v4f y1 = *(const v4f*)(p + 20);
    v8h lo, hi;
#pragma unroll
    for (int i = 0; i < 4; ++i) {
        lo[i] = (_Float16)x0[i]; lo[i + 4] = (_Float16)x1[i];
        hi[i] = (_Float16)y0[i]; hi[i + 4] = (_Float16)y1[i];
    }
    return pack16(lo, hi);
}

// B-fragment (32x16, f16, KxN) from K-contiguous (column-major) storage bt,
// bt = &B^T[n0][k0], row stride ld. Lane holds N=lane&15, K=[g*8, g*8+7] and [g*8+16, g*8+23].
static __device__ __forceinline__ v16h load_b_h(const _Float16* bt, int ld, int lane) {
    const int n = lane & 15, g = lane >> 4;
    const _Float16* p = bt + (size_t)n * ld + g * 8;
    return pack16(*(const v8h*)p, *(const v8h*)(p + 16));
}

static __device__ __forceinline__ v8f wmma16(v16h a, v16h b, v8f c) {
    return __builtin_amdgcn_wmma_f32_16x16x32_f16(false, a, false, b, (short)0, c, false, false);
}

// One 16-byte global->LDS transfer (async path uses ASYNCcnt-tracked DMA).
static __device__ __forceinline__ void copy16(const _Float16* g, _Float16* l) {
#if USE_ASYNC_STAGE
    __builtin_amdgcn_global_load_async_to_lds_b128(
        (AS_GLOBAL i128_t*)g, (AS_LDS i128_t*)l, 0, 0);
#else
    *(v8h*)l = *(const v8h*)g;
#endif
}

// ---------------------------------------------------------------------------
// Kernel 1: W[E][H] fp32 -> WT[3][H][E] fp16 (Wq scaled by 1/sqrt(E)).
// ---------------------------------------------------------------------------
__global__ void prep_weights_kernel(const float* __restrict__ Wk,
                                    const float* __restrict__ Wq,
                                    const float* __restrict__ Wv,
                                    _Float16* __restrict__ WT) {
    int idx = blockIdx.x * blockDim.x + threadIdx.x;
    const int per = HEADS * EMBED;
    if (idx >= 3 * per) return;
    int mat = idx / per;
    int r = idx - mat * per;
    int h = r / EMBED;
    int e = r - h * EMBED;
    const float* W = (mat == 0) ? Wk : (mat == 1) ? Wq : Wv;
    float v = W[e * HEADS + h];
    if (mat == 1) v *= 0.036084391824352f;  // 1/sqrt(768): fold score scale into Q
    WT[idx] = (_Float16)v;
}

// ---------------------------------------------------------------------------
// Kernel 2: QKV projection. One wave = one 16(M) x 16(N=H-slice) tile,
// K=768 in 24 WMMA steps. Q,K -> fp16 [b*s][64]; V -> transposed fp16 [b][h][s].
// ---------------------------------------------------------------------------
__global__ void __launch_bounds__(128)
project_kernel(const float* __restrict__ x,
               const _Float16* __restrict__ WT,
               _Float16* __restrict__ Kh,
               _Float16* __restrict__ Qh,
               _Float16* __restrict__ Vt) {
    const int tile = blockIdx.x / 3;     // 16-row M tile over B*S
    const int mat  = blockIdx.x % 3;     // 0=K, 1=Q, 2=V
    const int wave = threadIdx.x >> 5;
    const int lane = threadIdx.x & 31;
    const int hbase = wave * 16;         // N tile (H slice)
    const int sbase = tile * 16;         // global row in [0, B*S)

    const float* xrow = x + (size_t)sbase * EMBED;
    const _Float16* wt = WT + (size_t)mat * HEADS * EMBED + (size_t)hbase * EMBED;

    v8f c = {};
#pragma unroll 4
    for (int e = 0; e < EMBED; e += 32) {
        v16h a = load_a_f32(xrow + e, EMBED, lane);
        v16h b = load_b_h(wt + e, EMBED, lane);
        c = wmma16(a, b, c);
    }

    const int n = lane & 15, g = lane >> 4;
    if (mat == 2) {
        // Transposed store: element (m=r+8g, n) -> Vt[h=hbase+n][s=sbase+m],
        // contiguous in r -> single 16B store per lane.
        const int bidx = sbase >> 12;          // / SEQ
        const int srow = sbase & (SEQ - 1);
        v8h pk;
#pragma unroll
        for (int r = 0; r < 8; ++r) pk[r] = (_Float16)c[r];
        _Float16* dst = Vt + ((size_t)bidx * HEADS + hbase + n) * SEQ + srow + g * 8;
        *(v8h*)dst = pk;
    } else {
        _Float16* dst = (mat == 0) ? Kh : Qh;
#pragma unroll
        for (int r = 0; r < 8; ++r)
            dst[(size_t)(sbase + r + 8 * g) * HEADS + hbase + n] = (_Float16)c[r];
    }
}

// ---------------------------------------------------------------------------
// Kernel 3: flash attention. Block = 128 threads = 4 waves; each wave owns a
// 16-query tile. K/V stream in 64-key chunks, cooperatively staged into LDS
// double buffers (async DMA when available) so each chunk is fetched once per
// block instead of once per wave. Per chunk: 8 score WMMAs, fp32 online
// softmax, P staged wave-locally to LDS, 8 P*V WMMAs.
// ---------------------------------------------------------------------------
__global__ void __launch_bounds__(128)
attention_kernel(const _Float16* __restrict__ Qh,
                 const _Float16* __restrict__ Kh,
                 const _Float16* __restrict__ Vt,
                 float* __restrict__ out) {
    const int tid  = threadIdx.x;
    const int wave = tid >> 5;
    const int lane = tid & 31;
    const int qg   = blockIdx.x * 64 + wave * 16;   // global query row
    const int b    = qg >> 12;                      // / SEQ
    const int n    = lane & 15, g = lane >> 4;

    // Double-buffered K/V chunk staging + wave-local P staging (rows padded to
    // 72 halves = 144 B: 16B-aligned, bank-conflict friendly).
    __shared__ __align__(32) _Float16 Kl[2][64][72];   // [buf][key][h]
    __shared__ __align__(32) _Float16 Vl[2][64][72];   // [buf][h][key]
    __shared__ __align__(32) _Float16 Plds[4][16][72]; // [wave][q][key]

    // Q A-fragments (K-dim = h): h 0..31 and 32..63.
    const _Float16* Qb = Qh + (size_t)qg * HEADS;
    const v16h a_q0 = load_a_h(Qb, HEADS, lane);
    const v16h a_q1 = load_a_h(Qb + 32, HEADS, lane);

    const _Float16* Kb = Kh + ((size_t)b * SEQ) * HEADS;   // [4096][64]
    const _Float16* Vb = Vt + ((size_t)b * HEADS) * SEQ;   // [64][4096]

    float mrow[8], lrow[8];
#pragma unroll
    for (int r = 0; r < 8; ++r) { mrow[r] = -1e30f; lrow[r] = 0.0f; }
    v8f acc[4] = {};
    const v8f czero = {};

    // Stage one 64-key chunk: 8 KB K + 8 KB V, 8 x 16B per thread.
    auto stage = [&](int k0, int buf) {
#pragma unroll
        for (int j = 0; j < 4; ++j) {
            int e   = j * 128 + tid;     // 0..511
            int row = e >> 3;            // 0..63
            int col = (e & 7) * 8;       // half offset within 64
            copy16(Kb + (size_t)(k0 + row) * HEADS + col, &Kl[buf][row][col]);
            copy16(Vb + (size_t)row * SEQ + k0 + col,     &Vl[buf][row][col]);
        }
    };

    stage(0, 0);  // prologue

    for (int c = 0; c < NCHUNK; ++c) {
        const int buf = c & 1;
        if (c + 1 < NCHUNK) stage((c + 1) * 64, buf ^ 1);
#if USE_ASYNC_STAGE
        if (c + 1 < NCHUNK) __builtin_amdgcn_s_wait_asynccnt(8);  // chunk c done
        else                __builtin_amdgcn_s_wait_asynccnt(0);
#endif
        __syncthreads();  // chunk c staged by all waves

        // ---- scores: S[16 x 64] = Q * K^T (contraction over h=64) ----
        v8f s[4];
#pragma unroll
        for (int t = 0; t < 4; ++t) {
            const _Float16* kt = &Kl[buf][t * 16][0];
            v16h b0 = load_b_h(kt, 72, lane);        // h 0..31
            v16h b1 = load_b_h(kt + 32, 72, lane);   // h 32..63
            v8f cc = wmma16(a_q0, b0, czero);
            s[t] = wmma16(a_q1, b1, cc);
        }

        // ---- online softmax over the 64 keys ----
        float scale[8];
#pragma unroll
        for (int r = 0; r < 8; ++r) {
            float mx = mrow[r];
#pragma unroll
            for (int t = 0; t < 4; ++t) mx = fmaxf(mx, s[t][r]);
#pragma unroll
            for (int off = 1; off < 16; off <<= 1)
                mx = fmaxf(mx, __shfl_xor(mx, off, 16));
            scale[r] = __expf(mrow[r] - mx);
            mrow[r] = mx;
            lrow[r] *= scale[r];
        }
#pragma unroll
        for (int t = 0; t < 4; ++t)
#pragma unroll
            for (int r = 0; r < 8; ++r) s[t][r] = __expf(s[t][r] - mrow[r]);
#pragma unroll
        for (int r = 0; r < 8; ++r) {
            float sum = s[0][r] + s[1][r] + s[2][r] + s[3][r];
#pragma unroll
            for (int off = 1; off < 16; off <<= 1)
                sum += __shfl_xor(sum, off, 16);
            lrow[r] += sum;
        }
#pragma unroll
        for (int t = 0; t < 4; ++t)
#pragma unroll
            for (int r = 0; r < 8; ++r) acc[t][r] *= scale[r];

        // ---- stage P (f16) through LDS, reload as A-fragments ----
#pragma unroll
        for (int t = 0; t < 4; ++t)
#pragma unroll
            for (int r = 0; r < 8; ++r)
                Plds[wave][r + 8 * g][t * 16 + n] = (_Float16)s[t][r];
        asm volatile("s_wait_dscnt 0" ::: "memory");  // wave-local LDS RAW

        const _Float16* Pb = &Plds[wave][0][0];
        v16h a_p0 = load_a_h(Pb, 72, lane);        // keys k0 .. k0+31
        v16h a_p1 = load_a_h(Pb + 32, 72, lane);   // keys k0+32 .. k0+63

        // ---- acc += P * V (contraction over 64 keys) ----
#pragma unroll
        for (int t = 0; t < 4; ++t) {
            const _Float16* vt = &Vl[buf][t * 16][0];
            v16h bv0 = load_b_h(vt, 72, lane);
            v16h bv1 = load_b_h(vt + 32, 72, lane);
            acc[t] = wmma16(a_p0, bv0, acc[t]);
            acc[t] = wmma16(a_p1, bv1, acc[t]);
        }

        __syncthreads();  // all waves done reading buf before it is re-staged
    }

    // ---- normalize and store fp32 output ----
#pragma unroll
    for (int t = 0; t < 4; ++t) {
#pragma unroll
        for (int r = 0; r < 8; ++r) {
            float inv = 1.0f / lrow[r];
            out[(size_t)(qg + r + 8 * g) * HEADS + t * 16 + n] = acc[t][r] * inv;
        }
    }
}

// ---------------------------------------------------------------------------
extern "C" void kernel_launch(void* const* d_in, const int* in_sizes, int n_in,
                              void* d_out, int out_size, void* d_ws, size_t ws_size,
                              hipStream_t stream) {
    (void)in_sizes; (void)n_in; (void)out_size; (void)ws_size;
    const float* x  = (const float*)d_in[0];
    const float* Wk = (const float*)d_in[1];
    const float* Wq = (const float*)d_in[2];
    const float* Wv = (const float*)d_in[3];
    float* out = (float*)d_out;

    // Workspace layout (fp16): WT[3][64][768] | Qh | Kh | Vt (each B*S*H)
    char* ws = (char*)d_ws;
    const size_t szWT  = (size_t)3 * HEADS * EMBED * sizeof(_Float16);   // 288 KB
    const size_t szQKV = (size_t)BATCH * SEQ * HEADS * sizeof(_Float16); // 2 MB
    _Float16* WT = (_Float16*)ws;
    _Float16* Qh = (_Float16*)(ws + szWT);
    _Float16* Kh = (_Float16*)(ws + szWT + szQKV);
    _Float16* Vt = (_Float16*)(ws + szWT + 2 * szQKV);

    // 1) fp16 transposed weights (+ 1/sqrt(E) folded into Wq)
    {
        int total = 3 * HEADS * EMBED;
        prep_weights_kernel<<<(total + 255) / 256, 256, 0, stream>>>(Wk, Wq, Wv, WT);
    }
    // 2) QKV projection: (B*S/16) M-tiles x 3 matrices; 4 waves = 4 H-tiles
    {
        int mtiles = (BATCH * SEQ) / 16;   // 1024
        project_kernel<<<mtiles * 3, 128, 0, stream>>>(x, WT, Kh, Qh, Vt);
    }
    // 3) flash attention: 64 queries per block (4 waves x 16)
    {
        int blocks = (BATCH * SEQ) / 64;   // 256
        attention_kernel<<<blocks, 128, 0, stream>>>(Qh, Kh, Vt, out);
    }
}
